// CSplineBasic_15058155339814
// MI455X (gfx1250) — matrix operand order repaired and checked by reading, not compiled, not run on hardware
//
#include <hip/hip_runtime.h>

typedef __attribute__((ext_vector_type(16))) _Float16 v16h;
typedef __attribute__((ext_vector_type(8)))  _Float16 v8h;
typedef __attribute__((ext_vector_type(8)))  float    v8f;

#define BN    16384
#define CONDN 128
#define KNOTS 10

static __device__ __forceinline__ v8f wmma_f16(v16h a, v16h b, v8f c) {
  return __builtin_amdgcn_wmma_f32_16x16x32_f16(false, a, false, b, (short)0, c,
                                                false, false);
}

// Broadcast a scalar (the layer bias for this lane's N column) into a C
// accumulator. C layout has N in lanes, M in VGPRs, and bias depends only on
// N, so every element of this lane's accumulator gets the same value.
static __device__ __forceinline__ v8f cinit(float bias) {
  v8f c;
#pragma unroll
  for (int r = 0; r < 8; ++r) c[r] = bias;
  return c;
}

// Load WMMA A operand (16x32 f16) for one 32-wide K window from an LDS tile
// stored row-major with rowStride f16 elements per row.
// A layout: lane l<16 -> row=l, K = {0..7, 16..23}; lane l>=16 -> row=l-16,
// K = {8..15, 24..31}.
static __device__ __forceinline__ v16h ldA16(const _Float16* base, int rowStride,
                                             int row, int kc, int hf) {
  const _Float16* p = base + row * rowStride + kc * 32 + 8 * hf;
  v8h lo = *(const v8h*)p;          // K window +  {0..7}  (or {8..15})
  v8h hi = *(const v8h*)(p + 16);   // K window + {16..23} (or {24..31})
  v16h a;
#pragma unroll
  for (int i = 0; i < 8; ++i) { a[i] = lo[i]; a[i + 8] = hi[i]; }
  return a;
}

// Load WMMA B operand (32x16 f16) from an LDS weight image stored as
// W^T[n][k] (row stride kPad f16). B layout: lanes 0-15 hold K={0..7,16..23} at N=l,
// lanes 16-31 hold K={8..15,24..31} at N=l-16, matching the A layout.
static __device__ __forceinline__ v16h ldB16(const _Float16* base, int kPad,
                                             int lr, int t, int kc, int hf) {
  const _Float16* p = base + (lr + 16 * t) * kPad + kc * 32 + 8 * hf;
  v8h lo = *(const v8h*)p;
  v8h hi = *(const v8h*)(p + 16);
  v16h b;
#pragma unroll
  for (int i = 0; i < 8; ++i) { b[i] = lo[i]; b[i + 8] = hi[i]; }
  return b;
}

// ---------------------------------------------------------------------------
// Kernel 1: batch-constant values k_null = fm(null_token), k_v = g(sin(iv)).
// One wave of 32 threads, pure VALU (negligible work).
// ---------------------------------------------------------------------------
static __device__ float cstep(float* buf, float a, const float* W,
                              const float* bias, bool relu, int l) {
  buf[l] = a;
  __syncthreads();
  float acc = bias[l];
  for (int k = 0; k < 32; ++k) acc = fmaf(buf[k], W[k * 32 + l], acc);
  __syncthreads();
  return relu ? fmaxf(acc, 0.f) : acc;
}

__global__ void const_kernel(const float* faW0, const float* faB0,
                             const float* faW1, const float* faB1,
                             const float* faW2, const float* faB2,
                             const float* fmW0, const float* fmB0,
                             const float* fmW1, const float* fmB1,
                             const float* fmW2, const float* fmB2,
                             const float* null_token, const int* index_v,
                             float* csts) {
  __shared__ float buf[32];
  const int l = threadIdx.x;
  // k_v = g(sin(index_v))
  float s = sinf((float)index_v[0]);
  float a = fmaxf(fmaf(faW0[l], s, faB0[l]), 0.f);
  a = cstep(buf, a, faW1, faB1, true, l);
  a = cstep(buf, a, faW2, faB2, false, l);
  a = cstep(buf, a, fmW0, fmB0, true, l);
  a = cstep(buf, a, fmW1, fmB1, true, l);
  buf[l] = a;
  __syncthreads();
  if (l == 0) {
    float acc = fmB2[0];
    for (int k = 0; k < 32; ++k) acc = fmaf(buf[k], fmW2[k], acc);
    csts[1] = acc;
  }
  __syncthreads();
  // k_null = fm(null_token)
  float a2 = null_token[l];
  a2 = cstep(buf, a2, fmW0, fmB0, true, l);
  a2 = cstep(buf, a2, fmW1, fmB1, true, l);
  buf[l] = a2;
  __syncthreads();
  if (l == 0) {
    float acc = fmB2[0];
    for (int k = 0; k < 32; ++k) acc = fmaf(buf[k], fmW2[k], acc);
    csts[0] = acc;
  }
}

// ---------------------------------------------------------------------------
// Kernel 2: conditioner cp[b,j] = g(c[b,j]) over B*COND scalars.
// Dominant kernel: 9 WMMAs per 16-row tile, weights held in B-operand layout
// in registers, bias folded into the WMMA C accumulator, ReLU/f16-convert
// repack through a 1KB/wave LDS tile.
// ---------------------------------------------------------------------------
__global__ __launch_bounds__(256) void cond_mlp_kernel(
    const float* __restrict__ c,
    const float* __restrict__ faW0, const float* __restrict__ faB0,
    const float* __restrict__ faW1, const float* __restrict__ faB1,
    const float* __restrict__ faW2, const float* __restrict__ faB2,
    const float* __restrict__ fmW0, const float* __restrict__ fmB0,
    const float* __restrict__ fmW1, const float* __restrict__ fmB1,
    const float* __restrict__ fmW2, const float* __restrict__ fmB2,
    _Float16* __restrict__ cph) {
  __shared__ __align__(16) _Float16 actS[8][16 * 32];
  const int lane = threadIdx.x & 31;
  const int wave = threadIdx.x >> 5;
  const int hf = lane >> 4;   // half-wave id
  const int lr = lane & 15;

  // Stage the four 32x32 weight matrices (faW1, faW2, fmW0, fmW1) into
  // B-operand layout registers (f16) + per-lane bias values.
  const float* Wg[4] = {faW1, faW2, fmW0, fmW1};
  const float* Bg[4] = {faB1, faB2, fmB0, fmB1};
  v16h Bw[4][2];
  float vb[4][2];
#pragma unroll
  for (int l = 0; l < 4; ++l) {
#pragma unroll
    for (int t = 0; t < 2; ++t) {
      const int n = lr + 16 * t;
#pragma unroll
      for (int v = 0; v < 8; ++v) {
        const int k = 2 * v + 8 * hf + ((v >= 4) ? 8 : 0); // B layout K index
        Bw[l][t][2 * v]     = (_Float16)Wg[l][k * 32 + n];
        Bw[l][t][2 * v + 1] = (_Float16)Wg[l][(k + 1) * 32 + n];
      }
      vb[l][t] = Bg[l][n];
    }
  }
  // Final 32->1 dot as a B tile with fmW2 in column 0.
  v16h Bw2;
#pragma unroll
  for (int v = 0; v < 8; ++v) {
    const int k = 2 * v + 8 * hf + ((v >= 4) ? 8 : 0);
    Bw2[2 * v]     = (lr == 0) ? (_Float16)fmW2[k]     : (_Float16)0.f;
    Bw2[2 * v + 1] = (lr == 0) ? (_Float16)fmW2[k + 1] : (_Float16)0.f;
  }
  const float b2s = fmB2[0];

  // fa layer 0 (scalar -> 32) coefficients, ordered per the A-operand layout.
  float w0p[16], b0p[16];
#pragma unroll
  for (int v = 0; v < 8; ++v) {
    const int k = 2 * v + 8 * hf + ((v >= 4) ? 8 : 0); // A layout K index
    w0p[2 * v] = faW0[k];     w0p[2 * v + 1] = faW0[k + 1];
    b0p[2 * v] = faB0[k];     b0p[2 * v + 1] = faB0[k + 1];
  }

  _Float16* aw = &actS[wave][0];
  const int tile0 = (blockIdx.x * 8 + wave) * 16;   // 16 tiles per wave
  for (int tt = 0; tt < 16; ++tt) {
    const int base = (tile0 + tt) * 16;             // flat index into c
    const float s = c[base + lr];                   // this lane's row scalar
    // fa0: build A operand directly in registers (ReLU applied)
    v16h a;
#pragma unroll
    for (int i = 0; i < 16; ++i) {
      a[i] = (_Float16)fmaxf(fmaf(w0p[i], s, b0p[i]), 0.f);
    }
    // layers: fa1(relu), fa2(no relu), fm0(relu), fm1(relu)
    const bool reluF[4] = {true, false, true, true};
#pragma unroll
    for (int l = 0; l < 4; ++l) {
      v8f c0 = wmma_f16(a, Bw[l][0], cinit(vb[l][0]));  // bias pre-loaded in C
      v8f c1 = wmma_f16(a, Bw[l][1], cinit(vb[l][1]));
#pragma unroll
      for (int r = 0; r < 8; ++r) {
        const int M = r + 8 * hf;
        float y0 = c0[r];
        float y1 = c1[r];
        if (reluF[l]) { y0 = fmaxf(y0, 0.f); y1 = fmaxf(y1, 0.f); }
        aw[M * 32 + lr]      = (_Float16)y0;
        aw[M * 32 + lr + 16] = (_Float16)y1;
      }
      __asm__ volatile("s_wait_dscnt 0x0" ::: "memory");
      a = ldA16(aw, 32, lr, 0, hf);                 // repack C -> A layout
    }
    // fm2: 32 -> 1 via WMMA, result in column N=0 (lanes 0 and 16); b2s folded
    // into the accumulator (harmless in the unread columns).
    v8f cf = wmma_f16(a, Bw2, cinit(b2s));
    if (lr == 0) {
#pragma unroll
      for (int r = 0; r < 8; ++r)
        cph[base + r + 8 * hf] = (_Float16)cf[r];
    }
  }
}

// ---------------------------------------------------------------------------
// Rational-quadratic spline (scalar, per row).
// ---------------------------------------------------------------------------
static __device__ void spline_eval(const float* pv, float x, float& yOut,
                                   float& jaOut) {
  float ew[KNOTS], eh[KNOTS];
  float mw = pv[0], mh = pv[KNOTS];
#pragma unroll
  for (int i = 1; i < KNOTS; ++i) {
    mw = fmaxf(mw, pv[i]);
    mh = fmaxf(mh, pv[KNOTS + i]);
  }
  float se = 0.f, sh = 0.f;
#pragma unroll
  for (int i = 0; i < KNOTS; ++i) {
    ew[i] = expf(pv[i] - mw);         se += ew[i];
    eh[i] = expf(pv[KNOTS + i] - mh); sh += eh[i];
  }
  const float sw = 10.f / se, shs = 10.f / sh;   // softmax * 2*BINT
  float w[KNOTS + 1], h[KNOTS + 1], d[KNOTS + 1];
  w[0] = -5.f; h[0] = -5.f; d[0] = 1.f; d[KNOTS] = 1.f;
#pragma unroll
  for (int i = 0; i < KNOTS; ++i) {
    w[i + 1] = w[i] + ew[i] * sw;
    h[i + 1] = h[i] + eh[i] * shs;
  }
#pragma unroll
  for (int i = 1; i < KNOTS; ++i) {
    float z = pv[2 * KNOTS + (i - 1)];
    float sp = (z > 20.f) ? z : log1pf(expf(z));   // softplus
    d[i] = sp + 0.001f;
  }
  int idx = 0;
#pragma unroll
  for (int i = 0; i <= KNOTS; ++i) idx += (w[i] <= x) ? 1 : 0;
  idx = (idx < 1) ? 1 : ((idx > KNOTS) ? KNOTS : idx);
  float wl = 0, wr = 0, hl = 0, hr = 0, dl = 0, dr = 0;
#pragma unroll
  for (int i = 1; i <= KNOTS; ++i) {
    if (i == idx) { wl = w[i-1]; wr = w[i]; hl = h[i-1]; hr = h[i];
                    dl = d[i-1]; dr = d[i]; }
  }
  const float t = (x - wl) / (wr - wl);
  const float sl = (hr - hl) / (wr - wl);
  const float omt = 1.f - t;
  const float v1 = sl * t * t + dl * t * omt;
  const float v2 = sl + (dr + dl - 2.f * sl) * t * omt;
  yOut = hl + (v1 / v2) * (hr - hl);
  const float num = dr * t * t + 2.f * sl * t * omt + dl * omt * omt;
  jaOut = fabsf(sl * sl * num / (v2 * v2));
}

static __device__ __forceinline__ float getin(int g, int row,
                                              const _Float16* cph, float knull,
                                              float kv, int idxp, float xfirst) {
  if (g == 0) return xfirst;         // x11 (pass0) or x22 (pass1)
  if (g >= 130) return 0.f;          // K padding
  const int j = g - 1;
  if (j == 128) return kv;           // appended sin(index_v) column
  if (j == idxp || j == idxp + 64) return knull;
  return (float)cph[row * CONDN + j];
}

// ---------------------------------------------------------------------------
// Kernel 3: f1/f2 MLP (130->64->64->29 via WMMA) + spline. pass=0 computes
// x22/ja1 into workspace, pass=1 produces the final (x3, ja) outputs.
// ---------------------------------------------------------------------------
__global__ __launch_bounds__(256) void flow_mlp_kernel(
    const float* __restrict__ xg, const _Float16* __restrict__ cph,
    const float* __restrict__ csts, const int* __restrict__ idxpP,
    const float* __restrict__ W0g, const float* __restrict__ b0g,
    const float* __restrict__ W1g, const float* __restrict__ b1g,
    const float* __restrict__ W2g, const float* __restrict__ b2g,
    float* __restrict__ x22io, float* __restrict__ ja1io,
    float* __restrict__ outp, const int pass) {
  __shared__ __align__(16) _Float16 Wl0[64 * 160];   // W0^T [n][k], K pad 160
  __shared__ __align__(16) _Float16 Wl1[64 * 64];    // W1^T [n][k]
  __shared__ __align__(16) _Float16 Wl2[32 * 64];    // W2^T [n][k], N pad 32
  __shared__ __align__(16) _Float16 actS[8][16 * 64];

  const int tid = threadIdx.x;
  for (int i = tid; i < 64 * 160; i += 256) {
    const int n = i / 160, k = i - n * 160;
    Wl0[i] = (k < 130) ? (_Float16)W0g[k * 64 + n] : (_Float16)0.f;
  }
  for (int i = tid; i < 64 * 64; i += 256) {
    const int n = i >> 6, k = i & 63;
    Wl1[i] = (_Float16)W1g[k * 64 + n];
  }
  for (int i = tid; i < 32 * 64; i += 256) {
    const int n = i >> 6, k = i & 63;
    Wl2[i] = (n < 29) ? (_Float16)W2g[k * 29 + n] : (_Float16)0.f;
  }
  __syncthreads();

  const int lane = tid & 31, wave = tid >> 5;
  const int hf = lane >> 4, lr = lane & 15;
  const float knull = csts[0], kv = csts[1];
  const int idxp = idxpP[0];
  const int rowbase = (blockIdx.x * 8 + wave) * 16;
  const int row = rowbase + lr;
  const float xfirst = (pass == 0) ? xg[row * 2] : x22io[row];

  // Build the five layer-0 A chunks (K = 0..159) straight into A layout.
  v16h A0[5];
#pragma unroll
  for (int kc = 0; kc < 5; ++kc) {
#pragma unroll
    for (int v = 0; v < 8; ++v) {
      const int gk = kc * 32 + 2 * v + 8 * hf + ((v >= 4) ? 8 : 0);
      A0[kc][2 * v]     = (_Float16)getin(gk,     row, cph, knull, kv, idxp, xfirst);
      A0[kc][2 * v + 1] = (_Float16)getin(gk + 1, row, cph, knull, kv, idxp, xfirst);
    }
  }
  _Float16* aw = &actS[wave][0];

  // Layer 0: 130(pad 160) -> 64, bias in accumulator
#pragma unroll
  for (int t = 0; t < 4; ++t) {
    v8f C = cinit(b0g[lr + 16 * t]);
#pragma unroll
    for (int kc = 0; kc < 5; ++kc)
      C = wmma_f16(A0[kc], ldB16(Wl0, 160, lr, t, kc, hf), C);
#pragma unroll
    for (int r = 0; r < 8; ++r) {
      const int M = r + 8 * hf;
      aw[M * 64 + lr + 16 * t] = (_Float16)fmaxf(C[r], 0.f);
    }
  }
  __asm__ volatile("s_wait_dscnt 0x0" ::: "memory");

  // Layer 1: 64 -> 64 (A chunks fully loaded before stores -> single buffer)
  v16h A1a = ldA16(aw, 64, lr, 0, hf);
  v16h A1b = ldA16(aw, 64, lr, 1, hf);
#pragma unroll
  for (int t = 0; t < 4; ++t) {
    v8f C = cinit(b1g[lr + 16 * t]);
    C = wmma_f16(A1a, ldB16(Wl1, 64, lr, t, 0, hf), C);
    C = wmma_f16(A1b, ldB16(Wl1, 64, lr, t, 1, hf), C);
#pragma unroll
    for (int r = 0; r < 8; ++r) {
      const int M = r + 8 * hf;
      aw[M * 64 + lr + 16 * t] = (_Float16)fmaxf(C[r], 0.f);
    }
  }
  __asm__ volatile("s_wait_dscnt 0x0" ::: "memory");

  // Layer 2: 64 -> 29 (pad 32), f32 result into reused act region
  v16h A2a = ldA16(aw, 64, lr, 0, hf);
  v16h A2b = ldA16(aw, 64, lr, 1, hf);
  float* pbuf = (float*)aw;   // [16][32] f32 == same 2KB footprint
#pragma unroll
  for (int t = 0; t < 2; ++t) {
    const int n = lr + 16 * t;
    v8f C = cinit((n < 29) ? b2g[n] : 0.f);
    C = wmma_f16(A2a, ldB16(Wl2, 64, lr, t, 0, hf), C);
    C = wmma_f16(A2b, ldB16(Wl2, 64, lr, t, 1, hf), C);
#pragma unroll
    for (int r = 0; r < 8; ++r) {
      const int M = r + 8 * hf;
      pbuf[M * 32 + n] = C[r];            // no ReLU on last layer
    }
  }
  __asm__ volatile("s_wait_dscnt 0x0" ::: "memory");

  // Spline on lanes 0..15 (one row each).
  if (hf == 0) {
    float pv[29];
#pragma unroll
    for (int i = 0; i < 29; ++i) pv[i] = pbuf[lr * 32 + i];
    const float xin = (pass == 0) ? xg[row * 2 + 1] : xg[row * 2];
    float y, ja;
    spline_eval(pv, xin, y, ja);
    if (pass == 0) {
      x22io[row] = y;
      ja1io[row] = ja;
    } else {
      outp[row * 2]      = y;             // x31
      outp[row * 2 + 1]  = x22io[row];    // x22
      outp[2 * BN + row] = ja1io[row] * ja;
    }
  }
}

// ---------------------------------------------------------------------------
extern "C" void kernel_launch(void* const* d_in, const int* in_sizes, int n_in,
                              void* d_out, int out_size, void* d_ws,
                              size_t ws_size, hipStream_t stream) {
  (void)in_sizes; (void)n_in; (void)out_size; (void)ws_size;
  const float* xg         = (const float*)d_in[0];
  const float* cg         = (const float*)d_in[1];
  const int*   index_p    = (const int*)d_in[2];
  const int*   index_v    = (const int*)d_in[3];
  const float* null_token = (const float*)d_in[4];
  const float *f1W0 = (const float*)d_in[5],  *f1b0 = (const float*)d_in[6],
              *f1W1 = (const float*)d_in[7],  *f1b1 = (const float*)d_in[8],
              *f1W2 = (const float*)d_in[9],  *f1b2 = (const float*)d_in[10];
  const float *f2W0 = (const float*)d_in[11], *f2b0 = (const float*)d_in[12],
              *f2W1 = (const float*)d_in[13], *f2b1 = (const float*)d_in[14],
              *f2W2 = (const float*)d_in[15], *f2b2 = (const float*)d_in[16];
  const float *faW0 = (const float*)d_in[17], *fab0 = (const float*)d_in[18],
              *faW1 = (const float*)d_in[19], *fab1 = (const float*)d_in[20],
              *faW2 = (const float*)d_in[21], *fab2 = (const float*)d_in[22];
  const float *fmW0 = (const float*)d_in[23], *fmb0 = (const float*)d_in[24],
              *fmW1 = (const float*)d_in[25], *fmb1 = (const float*)d_in[26],
              *fmW2 = (const float*)d_in[27], *fmb2 = (const float*)d_in[28];

  // Workspace layout: cp (f16, B*COND) | consts(2 f32, padded) | x22 | ja1
  char* ws = (char*)d_ws;
  _Float16* cph = (_Float16*)ws;
  size_t off = (size_t)BN * CONDN * sizeof(_Float16);
  float* csts = (float*)(ws + off); off += 256;
  float* x22  = (float*)(ws + off); off += (size_t)BN * sizeof(float);
  float* ja1  = (float*)(ws + off);

  const_kernel<<<1, 32, 0, stream>>>(faW0, fab0, faW1, fab1, faW2, fab2,
                                     fmW0, fmb0, fmW1, fmb1, fmW2, fmb2,
                                     null_token, index_v, csts);
  // B*COND/16 = 131072 tiles; 16 tiles/wave, 8 waves/block -> 1024 blocks.
  cond_mlp_kernel<<<1024, 256, 0, stream>>>(cg, faW0, fab0, faW1, fab1, faW2,
                                            fab2, fmW0, fmb0, fmW1, fmb1, fmW2,
                                            fmb2, cph);
  // B/16 = 1024 row tiles; 8 waves/block -> 128 blocks, two passes.
  flow_mlp_kernel<<<128, 256, 0, stream>>>(xg, cph, csts, index_p,
                                           f1W0, f1b0, f1W1, f1b1, f1W2, f1b2,
                                           x22, ja1, (float*)d_out, 0);
  flow_mlp_kernel<<<128, 256, 0, stream>>>(xg, cph, csts, index_p,
                                           f2W0, f2b0, f2W1, f2b1, f2W2, f2b2,
                                           x22, ja1, (float*)d_out, 1);
}
